// Attention_11845519803093
// MI455X (gfx1250) — compile-verified, matrix operand order checked by reading, not run on hardware
//
#include <hip/hip_runtime.h>
#include <hip/hip_bf16.h>
#include <stdint.h>

// ---------------- problem constants ----------------
#define BATCH   16
#define SEQ     577
#define EDIM    768
#define HEADS   12
#define HDIM    64
#define QKVD    768          // HEADS*HDIM
#define ROWS    (BATCH*SEQ)  // 9232
#define TABR    30
#define MAXREL  14
#define SIDE    24
#define SCL     0.125f       // 64^-0.5

typedef __attribute__((ext_vector_type(16))) __bf16 v16bf;
typedef __attribute__((ext_vector_type(8)))  __bf16 bf16x8;
typedef __attribute__((ext_vector_type(8)))  float  v8f;
typedef __attribute__((ext_vector_type(4)))  int    v4i;

// ---------------- CDNA5 async LDS path (probe-guarded) ----------------
#if defined(__AMDGCN__) && __has_builtin(__builtin_amdgcn_global_load_async_to_lds_b128)
#define HAVE_ASYNC_LDS 1
#else
#define HAVE_ASYNC_LDS 0
#endif

#if HAVE_ASYNC_LDS
typedef __attribute__((address_space(1))) v4i as1_v4i;
typedef __attribute__((address_space(3))) v4i as3_v4i;
static __device__ __forceinline__ void async_b128(const void* g, void* l) {
  // GLOBAL_LOAD_ASYNC_TO_LDS_B128: 16B/lane, tracked by ASYNCcnt
  __builtin_amdgcn_global_load_async_to_lds_b128((as1_v4i*)g, (as3_v4i*)l, 0, 0);
}
#endif

static __device__ __forceinline__ void wait_async_le4() {
#if HAVE_ASYNC_LDS
#if __has_builtin(__builtin_amdgcn_s_wait_asynccnt)
  __builtin_amdgcn_s_wait_asynccnt(4);
#else
  asm volatile("s_wait_asynccnt 0x4" ::: "memory");
#endif
#endif
}
static __device__ __forceinline__ void wait_async_le0() {
#if HAVE_ASYNC_LDS
#if __has_builtin(__builtin_amdgcn_s_wait_asynccnt)
  __builtin_amdgcn_s_wait_asynccnt(0);
#else
  asm volatile("s_wait_asynccnt 0x0" ::: "memory");
#endif
#endif
}

// ---------------- helpers ----------------
static __device__ __forceinline__ v8f wmma_bf16(v16bf a, v16bf b, v8f c) {
  return __builtin_amdgcn_wmma_f32_16x16x32_bf16(false, a, false, b,
                                                 (short)0, c, false, false);
}

// 16-bit A/B fragment (16 rows x 32 K): per ISA layout, lane half selects
// K sub-ranges {half*8..+7} and {16+half*8..+7}; two 16B loads per lane.
static __device__ __forceinline__ v16bf load_frag(const __bf16* rowbase, int half) {
  union { v16bf v; bf16x8 h[2]; } u;
  u.h[0] = *reinterpret_cast<const bf16x8*>(rowbase + half * 8);
  u.h[1] = *reinterpret_cast<const bf16x8*>(rowbase + 16 + half * 8);
  return u.v;
}

static __device__ __forceinline__ float rmax16(float v) {
  v = fmaxf(v, __shfl_xor(v, 1));
  v = fmaxf(v, __shfl_xor(v, 2));
  v = fmaxf(v, __shfl_xor(v, 4));
  v = fmaxf(v, __shfl_xor(v, 8));
  return v;
}
static __device__ __forceinline__ float rsum16(float v) {
  v += __shfl_xor(v, 1);
  v += __shfl_xor(v, 2);
  v += __shfl_xor(v, 4);
  v += __shfl_xor(v, 8);
  return v;
}

static __device__ __forceinline__ void relidx(int q, int k, int& iv, int& ih) {
  if (q == 0 || k == 0) { iv = 0; ih = 0; return; }
  int qi = q - 1, ki = k - 1;
  int dv = ki / SIDE - qi / SIDE;
  int dh = ki % SIDE - qi % SIDE;
  dv = dv < -MAXREL ? -MAXREL : (dv > MAXREL ? MAXREL : dv);
  dh = dh < -MAXREL ? -MAXREL : (dh > MAXREL ? MAXREL : dh);
  iv = dv + MAXREL + 1;
  ih = dh + MAXREL + 1;
}

// ---------------- f32 -> bf16 conversion ----------------
__global__ void cvt_bf16_kernel(const float* __restrict__ in,
                                __bf16* __restrict__ out, int n) {
  int i = blockIdx.x * blockDim.x + threadIdx.x;
  if (i < n) out[i] = (__bf16)in[i];
}

// ---------------- tiled bf16 WMMA GEMM: C[M,N] = A[M,K] * B[N,K]^T ----------
// 256 threads = 8 waves; tile 128x128; wave grid 4x2; wave -> 32x64 (2x4 frags)
// Double-buffered LDS; producer is GLOBAL_LOAD_ASYNC_TO_LDS_B128 when present.
__global__ __launch_bounds__(256) void gemm_bf16_kernel(
    const __bf16* __restrict__ A, const __bf16* __restrict__ B,
    float* __restrict__ outF, __bf16* __restrict__ outBF,
    const float* __restrict__ bias, int M, int N, int K) {
  __shared__ __bf16 As[2][128 * 40];  // stride 40 (80B) keeps 16B alignment
  __shared__ __bf16 Bs[2][128 * 40];

  const int tid  = threadIdx.x;
  const int lane = tid & 31;
  const int half = lane >> 4;
  const int col  = lane & 15;
  const int w    = tid >> 5;
  const int wm   = w >> 1;            // 0..3
  const int wn   = w & 1;             // 0..1
  const int m0   = blockIdx.x * 128;
  const int n0   = blockIdx.y * 128;

  v8f acc[2][4];
#pragma unroll
  for (int i = 0; i < 2; i++)
#pragma unroll
    for (int j = 0; j < 4; j++)
#pragma unroll
      for (int e = 0; e < 8; e++) acc[i][j][e] = 0.f;

  const int r   = tid >> 1;   // 0..127: LDS tile row
  const int seg = tid & 1;    // 16-element K segment
  // OOB rows clamp to the last valid row; their C rows are masked at store.
  const int ar = (m0 + r < M) ? m0 + r : M - 1;
  const int br = (n0 + r < N) ? n0 + r : N - 1;
  const __bf16* gArow = A + (size_t)ar * K + seg * 16;
  const __bf16* gBrow = B + (size_t)br * K + seg * 16;
  __bf16* lA = &As[0][r * 40 + seg * 16];
  __bf16* lB = &Bs[0][r * 40 + seg * 16];

  auto stage = [&](int kt, int bufi) {
    const __bf16* ga = gArow + kt * 32;
    const __bf16* gb = gBrow + kt * 32;
    __bf16* la = lA + bufi * (128 * 40);
    __bf16* lb = lB + bufi * (128 * 40);
#if HAVE_ASYNC_LDS
    async_b128(ga,     la);
    async_b128(ga + 8, la + 8);
    async_b128(gb,     lb);
    async_b128(gb + 8, lb + 8);
#else
    const bf16x8* pa = reinterpret_cast<const bf16x8*>(ga);
    bf16x8 a0 = pa[0], a1 = pa[1];
    const bf16x8* pb = reinterpret_cast<const bf16x8*>(gb);
    bf16x8 b0 = pb[0], b1 = pb[1];
    *reinterpret_cast<bf16x8*>(la)     = a0;
    *reinterpret_cast<bf16x8*>(la + 8) = a1;
    *reinterpret_cast<bf16x8*>(lb)     = b0;
    *reinterpret_cast<bf16x8*>(lb + 8) = b1;
#endif
  };

  const int KT = K >> 5;  // K/32
  stage(0, 0);
  for (int kt = 0; kt < KT; kt++) {
    const int cur = kt & 1;
    if (kt + 1 < KT) {
      stage(kt + 1, cur ^ 1);
      if (kt + 2 < KT) {  // warm L2 two tiles ahead -> global_prefetch_b8
        __builtin_prefetch(gArow + (kt + 2) * 32, 0, 1);
        __builtin_prefetch(gBrow + (kt + 2) * 32, 0, 1);
      }
      wait_async_le4();   // previous tile landed; next tile still in flight
    } else {
      wait_async_le0();
    }
    __syncthreads();

    v16bf af[2], bfr[4];
#pragma unroll
    for (int i = 0; i < 2; i++)
      af[i] = load_frag(&As[cur][(wm * 32 + i * 16 + col) * 40], half);
#pragma unroll
    for (int j = 0; j < 4; j++)
      bfr[j] = load_frag(&Bs[cur][(wn * 64 + j * 16 + col) * 40], half);
#pragma unroll
    for (int i = 0; i < 2; i++)
#pragma unroll
      for (int j = 0; j < 4; j++)
        acc[i][j] = wmma_bf16(af[i], bfr[j], acc[i][j]);

    __syncthreads();  // protect buf `cur` before it is re-staged at kt+1
  }

  // epilogue; C layout: VGPR e -> M = e + half*8, N = col
#pragma unroll
  for (int i = 0; i < 2; i++)
#pragma unroll
    for (int j = 0; j < 4; j++)
#pragma unroll
      for (int e = 0; e < 8; e++) {
        int row = m0 + wm * 32 + i * 16 + half * 8 + e;
        int cl  = n0 + wn * 64 + j * 16 + col;
        if (row < M) {
          float v = acc[i][j][e];
          if (bias)  v += bias[cl];
          if (outF)  outF[(size_t)row * N + cl]  = v;
          if (outBF) outBF[(size_t)row * N + cl] = (__bf16)v;
        }
      }
}

// ---------------- fused attention with relative-position bias ---------------
// grid: (B*HEADS, ceil(SEQ/64)); 128 threads = 4 waves; wave owns 16 queries.
__global__ __launch_bounds__(128) void attn_kernel(
    const __bf16* __restrict__ qkv,   // [ROWS, 3*QKVD]
    const __bf16* __restrict__ tkv,   // [30,64] bf16
    const __bf16* __restrict__ tkh,   // [30,64] bf16
    const float*  __restrict__ tvv,   // [30,64] f32
    const float*  __restrict__ tvh,   // [30,64] f32
    __bf16* __restrict__ outbf) {     // [ROWS, EDIM]
  __shared__ __bf16 Vt[64 * 32];            // V tile transposed [d][k]
  __shared__ __bf16 pbuf[4][16 * 32];       // per-wave P tile
  __shared__ float  sv[4][16][TABR];
  __shared__ float  shh[4][16][TABR];
  __shared__ float  wv[4][16][TABR];
  __shared__ float  wh[4][16][TABR];
  __shared__ float  frow[4][16];
  __shared__ float  vbias[4][16 * HDIM];

  const int tid  = threadIdx.x;
  const int lane = tid & 31;
  const int half = lane >> 4;
  const int col  = lane & 15;
  const int wq   = tid >> 5;
  const int bh   = blockIdx.x;
  const int b    = bh / HEADS;
  const int h    = bh % HEADS;
  const int q0   = blockIdx.y * 64 + wq * 16;
  const size_t rs = 3 * QKVD;
  const __bf16* base = qkv + (size_t)b * SEQ * rs;

  // zero the value-bias bins
  for (int i = tid; i < 4 * 16 * TABR; i += 128) {
    ((float*)wv)[i] = 0.f;
    ((float*)wh)[i] = 0.f;
  }

  // sv/sh: q-row dotted with the 30 K-table rows (split 15/15 across halves)
  {
    int q = q0 + col; if (q > SEQ - 1) q = SEQ - 1;
    const __bf16* qr = base + (size_t)q * rs + h * HDIM;  // Q slab offset 0
    int tbase = half * 15;
    for (int tt = 0; tt < 15; tt++) {
      int t = tbase + tt;
      float s1 = 0.f, s2 = 0.f;
      for (int d = 0; d < HDIM; d++) {
        float qd = (float)qr[d];
        s1 += qd * (float)tkv[t * HDIM + d];
        s2 += qd * (float)tkh[t * HDIM + d];
      }
      sv[wq][col][t]  = s1;
      shh[wq][col][t] = s2;
    }
  }

  // Q A-fragments (16 x 64 -> two 16x32 fragments)
  v16bf qa[2];
  {
    int q = q0 + col; if (q > SEQ - 1) q = SEQ - 1;
    const __bf16* qr = base + (size_t)q * rs + h * HDIM;
    qa[0] = load_frag(qr, half);
    qa[1] = load_frag(qr + 32, half);
  }

  v8f o[4];
#pragma unroll
  for (int n = 0; n < 4; n++)
#pragma unroll
    for (int e = 0; e < 8; e++) o[n][e] = 0.f;
  float m[8], l[8];
#pragma unroll
  for (int e = 0; e < 8; e++) { m[e] = -3e38f; l[e] = 0.f; }

  const int nsteps = (SEQ + 31) / 32;  // 19
  for (int ks = 0; ks < nsteps; ks++) {
    int kb = ks * 32;
    __syncthreads();
    // cooperatively stage V tile transposed: Vt[d][kk]
    {
      int kk = tid & 31;
      int dg = tid >> 5;  // 0..3 -> d groups of 16
      int key = kb + kk; if (key > SEQ - 1) key = SEQ - 1;
      const __bf16* vr = base + (size_t)key * rs + 2 * QKVD + h * HDIM + dg * 16;
      const bf16x8* pv = reinterpret_cast<const bf16x8*>(vr);
      bf16x8 v0 = pv[0], v1 = pv[1];
#pragma unroll
      for (int i = 0; i < 8; i++) Vt[(dg * 16 + i) * 32 + kk] = v0[i];
#pragma unroll
      for (int i = 0; i < 8; i++) Vt[(dg * 16 + 8 + i) * 32 + kk] = v1[i];
    }
    __syncthreads();

    // S = Q * K^T for two 16-key subtiles (K-fragments straight from global)
    v8f S[2];
#pragma unroll
    for (int sub = 0; sub < 2; sub++) {
      int key = kb + sub * 16 + col; if (key > SEQ - 1) key = SEQ - 1;
      const __bf16* kr = base + (size_t)key * rs + QKVD + h * HDIM;
      v16bf kb0 = load_frag(kr, half);
      v16bf kb1 = load_frag(kr + 32, half);
      v8f z;
#pragma unroll
      for (int e = 0; e < 8; e++) z[e] = 0.f;
      z = wmma_bf16(qa[0], kb0, z);
      z = wmma_bf16(qa[1], kb1, z);
      S[sub] = z;
    }

    // bias + mask (store pre-softmax scores in p[][])
    float p[2][8];
#pragma unroll
    for (int e = 0; e < 8; e++) {
      int row = half * 8 + e;
      int q   = q0 + row;
      int qq  = q > SEQ - 1 ? SEQ - 1 : q;
#pragma unroll
      for (int sub = 0; sub < 2; sub++) {
        int key = kb + sub * 16 + col;
        int kk2 = key > SEQ - 1 ? SEQ - 1 : key;
        int iv, ih;
        relidx(qq, kk2, iv, ih);
        float s = (S[sub][e] + sv[wq][row][iv] + shh[wq][row][ih]) * SCL;
        if (key >= SEQ || q >= SEQ) s = -1e30f;
        p[sub][e] = s;
      }
    }

    // online softmax row update
    float fsc[8];
#pragma unroll
    for (int e = 0; e < 8; e++) {
      float rm = rmax16(fmaxf(p[0][e], p[1][e]));
      float mn = fmaxf(m[e], rm);
      float fs = __expf(m[e] - mn);
      m[e] = mn;
      float p0 = __expf(p[0][e] - mn);
      float p1 = __expf(p[1][e] - mn);
      p[0][e] = p0; p[1][e] = p1;
      l[e] = l[e] * fs + rsum16(p0 + p1);
      fsc[e] = fs;
#pragma unroll
      for (int n = 0; n < 4; n++) o[n][e] *= fs;
    }
    // publish per-row rescale, rescale value-bias bins (1 lane per row)
    if (col == 0) {
#pragma unroll
      for (int e = 0; e < 8; e++) frow[wq][half * 8 + e] = fsc[e];
    }
    if (col < 8) {
      int row = half * 8 + (lane & 7);
      float fs = frow[wq][row];
      for (int t = 0; t < TABR; t++) {
        wv[wq][row][t] *= fs;
        wh[wq][row][t] *= fs;
      }
    }

    // accumulate bins + write P tile (bf16) for the PV WMMA
#pragma unroll
    for (int e = 0; e < 8; e++) {
      int row = half * 8 + e;
      int q   = q0 + row; if (q > SEQ - 1) q = SEQ - 1;
#pragma unroll
      for (int sub = 0; sub < 2; sub++) {
        int key = kb + sub * 16 + col; if (key > SEQ - 1) key = SEQ - 1;
        int iv, ih;
        relidx(q, key, iv, ih);
        float pval = p[sub][e];
        atomicAdd(&wv[wq][row][iv], pval);
        atomicAdd(&wh[wq][row][ih], pval);
        pbuf[wq][row * 32 + sub * 16 + col] = (__bf16)pval;
      }
    }

    // O += P * V  (A = P from LDS, B = V^T tile from LDS)
    {
      v16bf pa = load_frag(&pbuf[wq][col * 32], half);
#pragma unroll
      for (int n = 0; n < 4; n++) {
        v16bf vb = load_frag(&Vt[(n * 16 + col) * 32], half);
        o[n] = wmma_bf16(pa, vb, o[n]);
      }
    }
  }

  // value-bias: vbias[r][d] = sum_t wv[r][t]*tvv[t][d] + wh[r][t]*tvh[t][d]
  for (int idx = lane; idx < 16 * HDIM; idx += 32) {
    int r = idx >> 6, d = idx & 63;
    float s = 0.f;
    for (int t = 0; t < TABR; t++)
      s += wv[wq][r][t] * tvv[t * HDIM + d] + wh[wq][r][t] * tvh[t * HDIM + d];
    vbias[wq][idx] = s;
  }

  // normalize and store bf16 in [b, n, h*64+d] layout for the proj GEMM
#pragma unroll
  for (int e = 0; e < 8; e++) {
    int row = half * 8 + e;
    int q   = q0 + row;
    if (q < SEQ) {
      float inv = 1.f / l[e];
#pragma unroll
      for (int n = 0; n < 4; n++) {
        int d = n * 16 + col;
        float v = (o[n][e] + vbias[wq][row * HDIM + d]) * inv;
        outbf[(size_t)(b * SEQ + q) * EDIM + h * HDIM + d] = (__bf16)v;
      }
    }
  }
}

// ---------------- launch ----------------
extern "C" void kernel_launch(void* const* d_in, const int* in_sizes, int n_in,
                              void* d_out, int out_size, void* d_ws, size_t ws_size,
                              hipStream_t stream) {
  const float* x      = (const float*)d_in[0];
  const float* qkv_w  = (const float*)d_in[1];
  const float* proj_w = (const float*)d_in[2];
  const float* proj_b = (const float*)d_in[3];
  const float* tab_kv = (const float*)d_in[4];
  const float* tab_kh = (const float*)d_in[5];
  const float* tab_vv = (const float*)d_in[6];
  const float* tab_vh = (const float*)d_in[7];
  float* out = (float*)d_out;

  char*  ws  = (char*)d_ws;
  size_t off = 0;
  auto alloc = [&](size_t bytes) -> void* {
    void* p = ws + off;
    off += (bytes + 255) & ~(size_t)255;
    return p;
  };
  __bf16* x_bf     = (__bf16*)alloc((size_t)ROWS * EDIM * 2);
  __bf16* qkvw_bf  = (__bf16*)alloc((size_t)3 * QKVD * EDIM * 2);
  __bf16* projw_bf = (__bf16*)alloc((size_t)EDIM * EDIM * 2);
  __bf16* tkv_bf   = (__bf16*)alloc((size_t)TABR * HDIM * 2);
  __bf16* tkh_bf   = (__bf16*)alloc((size_t)TABR * HDIM * 2);
  __bf16* qkv_bf   = (__bf16*)alloc((size_t)ROWS * 3 * QKVD * 2);
  __bf16* attn_bf  = (__bf16*)alloc((size_t)ROWS * EDIM * 2);

  auto cvt = [&](const float* src, __bf16* dst, int n) {
    cvt_bf16_kernel<<<(n + 255) / 256, 256, 0, stream>>>(src, dst, n);
  };
  cvt(x, x_bf, ROWS * EDIM);
  cvt(qkv_w, qkvw_bf, 3 * QKVD * EDIM);
  cvt(proj_w, projw_bf, EDIM * EDIM);
  cvt(tab_kv, tkv_bf, TABR * HDIM);
  cvt(tab_kh, tkh_bf, TABR * HDIM);

  // QKV GEMM: [9232,768] x [2304,768]^T -> bf16 [9232,2304]
  gemm_bf16_kernel<<<dim3((ROWS + 127) / 128, (3 * QKVD) / 128), 256, 0, stream>>>(
      x_bf, qkvw_bf, nullptr, qkv_bf, nullptr, ROWS, 3 * QKVD, EDIM);

  // fused attention
  attn_kernel<<<dim3(BATCH * HEADS, (SEQ + 63) / 64), 128, 0, stream>>>(
      qkv_bf, tkv_bf, tkh_bf, tab_vv, tab_vh, attn_bf);

  // proj GEMM: [9232,768] x [768,768]^T + bias -> f32 out
  gemm_bf16_kernel<<<dim3((ROWS + 127) / 128, EDIM / 128), 256, 0, stream>>>(
      attn_bf, projw_bf, out, nullptr, proj_b, ROWS, EDIM, EDIM);
}
